// CrystalEncoder_609885356788
// MI455X (gfx1250) — matrix operand order repaired and checked by reading, not compiled, not run on hardware
//
#include <hip/hip_runtime.h>
#include <hip/hip_bf16.h>

typedef __bf16 bf16;
typedef __attribute__((ext_vector_type(16))) __bf16 v16bf;
typedef __attribute__((ext_vector_type(8)))  __bf16 v8bf;
typedef __attribute__((ext_vector_type(8)))  float  v8f;

constexpr int BG     = 64;            // graphs
constexpr int PTS    = 256;           // points per graph
constexpr int NN     = BG * PTS;      // 16384 nodes
constexpr int DEG    = 16;
constexpr int EE     = NN * DEG;      // 262144 edges
constexpr int F_ATOM = 98;

// ---------------------------------------------------------------------------
// Per-graph NormalizeScale: center by mean, scale so max |coord| ~= 1.
// One block (256 threads) per graph.
// ---------------------------------------------------------------------------
__global__ void __launch_bounds__(256) k_normalize(const float* __restrict__ pos,
                                                   float* __restrict__ posn) {
  __shared__ float sx[256], sy[256], sz[256], sm[256];
  const int g = blockIdx.x, t = threadIdx.x;
  const int i = g * PTS + t;
  float x = pos[i * 3 + 0], y = pos[i * 3 + 1], z = pos[i * 3 + 2];
  sx[t] = x; sy[t] = y; sz[t] = z;
  __syncthreads();
  for (int s = 128; s > 0; s >>= 1) {
    if (t < s) { sx[t] += sx[t + s]; sy[t] += sy[t + s]; sz[t] += sz[t + s]; }
    __syncthreads();
  }
  const float inv = 1.0f / PTS;
  x -= sx[0] * inv; y -= sy[0] * inv; z -= sz[0] * inv;
  __syncthreads();
  sm[t] = fmaxf(fabsf(x), fmaxf(fabsf(y), fabsf(z)));
  __syncthreads();
  for (int s = 128; s > 0; s >>= 1) {
    if (t < s) sm[t] = fmaxf(sm[t], sm[t + s]);
    __syncthreads();
  }
  const float sc = 0.999999f / sm[0];
  posn[i * 3 + 0] = x * sc; posn[i * 3 + 1] = y * sc; posn[i * 3 + 2] = z * sc;
}

// ---------------------------------------------------------------------------
// Weight packing: W f32 [K,N] row-major -> bf16 fragments.
// packed[(kt*N + n)*32 + q] = W[kt*32 + q, n]  (zero-padded K -> Kp)
// so a lane's 16 B-fragment elements (K = kt*32 + half*8 + {0..7, 16..23}, fixed n)
// are two 16-byte runs starting 32 bytes apart.
// ---------------------------------------------------------------------------
__global__ void __launch_bounds__(256) k_pack_w(const float* __restrict__ W,
                                                bf16* __restrict__ out,
                                                int K, int Kp, int Ncols) {
  size_t idx = (size_t)blockIdx.x * 256 + threadIdx.x;
  size_t total = (size_t)Kp * Ncols;
  if (idx >= total) return;
  int kt  = (int)(idx / ((size_t)Ncols * 32));
  int rem = (int)(idx - (size_t)kt * Ncols * 32);
  int n   = rem >> 5;
  int q   = rem & 31;
  int k   = kt * 32 + q;
  float v = (k < K) ? W[(size_t)k * Ncols + n] : 0.0f;
  out[idx] = (bf16)v;
}

// ---------------------------------------------------------------------------
// WMMA GEMM: Y[M,N] = act(A[M,Kp] @ W + bias). A bf16 row-major (K padded to
// Kp, mult of 32), W packed as above, f32 accumulate, bf16 output.
// One wave computes a 32x64 strip: 2 A-fragments x 4 B-fragments = 8
// v_wmma_f32_16x16x32_bf16 per K-step, with register double-buffering so the
// next K-step's fragment loads overlap the current step's matrix ops.
// ---------------------------------------------------------------------------
__device__ __forceinline__ v16bf cat8(v8bf lo, v8bf hi) {
  return __builtin_shufflevector(lo, hi,
      0, 1, 2, 3, 4, 5, 6, 7, 8, 9, 10, 11, 12, 13, 14, 15);
}

__global__ void __launch_bounds__(256) k_wmma_gemm(const bf16* __restrict__ A,
                                                   const bf16* __restrict__ Wp,
                                                   const float* __restrict__ bias,
                                                   bf16* __restrict__ Y,
                                                   int M, int Kp, int Ncols,
                                                   int do_relu) {
  const int wave    = (blockIdx.x * 256 + threadIdx.x) >> 5;
  const int lane    = threadIdx.x & 31;
  const int tiles_n = Ncols >> 6;
  const int tm      = wave / tiles_n;
  const int tn      = wave - tm * tiles_n;
  if (tm >= (M >> 5)) return;

  const int mrow = lane & 15;   // A row within 16-tile / output col within n-tile
  const int half = lane >> 4;   // lane-half selects K sub-range

  // A fragment (ISA 16-bit A 16x32): lane holds K = kt*32 + {half*8+0..7, 16+half*8+0..7}
  const bf16* arow0 = A + (size_t)(tm * 32 + mrow) * Kp + half * 8;
  const bf16* arow1 = arow0 + (size_t)16 * Kp;
  // B fragment base: n = tn*64 + j*16 + mrow, 32 contiguous bf16 per (kt,n)
  const bf16* wbase = Wp + ((size_t)(tn * 64 + mrow) << 5) + (half << 3);

  auto loadA = [&](const bf16* p, int kt) -> v16bf {
    v8bf lo = *(const v8bf*)(p + kt * 32);
    v8bf hi = *(const v8bf*)(p + kt * 32 + 16);
    return cat8(lo, hi);
  };
  auto loadB = [&](int kt, int j) -> v16bf {
    const bf16* bp = wbase + (((size_t)kt * Ncols) << 5) + ((size_t)j << 9);
    v8bf lo = *(const v8bf*)(bp);
    v8bf hi = *(const v8bf*)(bp + 16);
    return cat8(lo, hi);
  };

  v8f acc[2][4] = {};

  const int nkt = Kp >> 5;
  // Prologue: fragments for kt = 0
  v16bf a0 = loadA(arow0, 0);
  v16bf a1 = loadA(arow1, 0);
  v16bf b0 = loadB(0, 0), b1 = loadB(0, 1), b2 = loadB(0, 2), b3 = loadB(0, 3);

  for (int kt = 0; kt < nkt; ++kt) {
    // Prefetch next K-step fragments before consuming current ones.
    v16bf na0 = a0, na1 = a1, nb0 = b0, nb1 = b1, nb2 = b2, nb3 = b3;
    if (kt + 1 < nkt) {
      na0 = loadA(arow0, kt + 1);
      na1 = loadA(arow1, kt + 1);
      nb0 = loadB(kt + 1, 0);
      nb1 = loadB(kt + 1, 1);
      nb2 = loadB(kt + 1, 2);
      nb3 = loadB(kt + 1, 3);
    }
    acc[0][0] = __builtin_amdgcn_wmma_f32_16x16x32_bf16(false, a0, false, b0, (short)0, acc[0][0], false, false);
    acc[1][0] = __builtin_amdgcn_wmma_f32_16x16x32_bf16(false, a1, false, b0, (short)0, acc[1][0], false, false);
    acc[0][1] = __builtin_amdgcn_wmma_f32_16x16x32_bf16(false, a0, false, b1, (short)0, acc[0][1], false, false);
    acc[1][1] = __builtin_amdgcn_wmma_f32_16x16x32_bf16(false, a1, false, b1, (short)0, acc[1][1], false, false);
    acc[0][2] = __builtin_amdgcn_wmma_f32_16x16x32_bf16(false, a0, false, b2, (short)0, acc[0][2], false, false);
    acc[1][2] = __builtin_amdgcn_wmma_f32_16x16x32_bf16(false, a1, false, b2, (short)0, acc[1][2], false, false);
    acc[0][3] = __builtin_amdgcn_wmma_f32_16x16x32_bf16(false, a0, false, b3, (short)0, acc[0][3], false, false);
    acc[1][3] = __builtin_amdgcn_wmma_f32_16x16x32_bf16(false, a1, false, b3, (short)0, acc[1][3], false, false);
    a0 = na0; a1 = na1; b0 = nb0; b1 = nb1; b2 = nb2; b3 = nb3;
  }

  // C/D layout: lane col n = lane&15, rows r + (half ? 8 : 0)
#pragma unroll
  for (int h = 0; h < 2; ++h) {
    const int rbase = tm * 32 + h * 16 + half * 8;
#pragma unroll
    for (int j = 0; j < 4; ++j) {
      int n = tn * 64 + j * 16 + mrow;
      float bb = bias[n];
#pragma unroll
      for (int r = 0; r < 8; ++r) {
        float v = acc[h][j][r] + bb;
        if (do_relu) v = v > 0.0f ? v : 0.0f;
        Y[(size_t)(rbase + r) * Ncols + n] = (bf16)v;
      }
    }
  }
}

// ---------------------------------------------------------------------------
// Per-column sum / sum-of-squares over M rows (BatchNorm batch stats).
// One block per column; no atomics, no pre-zeroing needed.
// ---------------------------------------------------------------------------
__global__ void __launch_bounds__(256) k_col_stats(const bf16* __restrict__ Y,
                                                   float* __restrict__ sum,
                                                   float* __restrict__ sumsq,
                                                   int M, int Ncols) {
  const int n = blockIdx.x, t = threadIdx.x;
  float s = 0.0f, q = 0.0f;
  for (int r = t; r < M; r += 256) {
    float v = (float)Y[(size_t)r * Ncols + n];
    s += v; q += v * v;
  }
  __shared__ float ss[256], qq[256];
  ss[t] = s; qq[t] = q;
  __syncthreads();
  for (int k = 128; k > 0; k >>= 1) {
    if (t < k) { ss[t] += ss[t + k]; qq[t] += qq[t + k]; }
    __syncthreads();
  }
  if (t == 0) { sum[n] = ss[0]; sumsq[n] = qq[0]; }
}

// ---------------------------------------------------------------------------
// BN apply (training-mode, biased var, eps=1e-5), in place, optional ReLU
// after normalization (head layers).
// ---------------------------------------------------------------------------
__global__ void __launch_bounds__(256) k_bn_apply(bf16* __restrict__ Y,
                                                  const float* __restrict__ sum,
                                                  const float* __restrict__ sumsq,
                                                  const float* __restrict__ gamma,
                                                  const float* __restrict__ beta,
                                                  int M, int Ncols, int relu_after) {
  size_t idx = (size_t)blockIdx.x * 256 + threadIdx.x;
  size_t total = (size_t)M * Ncols;
  if (idx >= total) return;
  int n = (int)(idx % Ncols);
  float invM = 1.0f / (float)M;
  float mu  = sum[n] * invM;
  float var = sumsq[n] * invM - mu * mu;
  float sc  = gamma[n] * rsqrtf(var + 1e-5f);
  float sh  = beta[n] - mu * sc;
  float v = (float)Y[idx] * sc + sh;
  if (relu_after) v = fmaxf(v, 0.0f);
  Y[idx] = (bf16)v;
}

// ---------------------------------------------------------------------------
// Edge feature gathers (zero-pad K to multiple of 32 for WMMA).
// ---------------------------------------------------------------------------
__global__ void __launch_bounds__(256) k_gather1(const float* __restrict__ x,
                                                 const float* __restrict__ posn,
                                                 const int* __restrict__ src,
                                                 const int* __restrict__ dst,
                                                 bf16* __restrict__ out) {
  size_t idx = (size_t)blockIdx.x * 256 + threadIdx.x;
  if (idx >= (size_t)EE * 128) return;
  int e = (int)(idx >> 7);
  int c = (int)(idx & 127);
  float v = 0.0f;
  if (c < F_ATOM) {
    v = x[(size_t)src[e] * F_ATOM + c];
  } else if (c < F_ATOM + 3) {
    int d = c - F_ATOM;
    v = posn[(size_t)src[e] * 3 + d] - posn[(size_t)dst[e] * 3 + d];
  }
  out[idx] = (bf16)v;
}

__global__ void __launch_bounds__(256) k_gather2(const bf16* __restrict__ h1,
                                                 const float* __restrict__ posn,
                                                 const int* __restrict__ src,
                                                 const int* __restrict__ dst,
                                                 bf16* __restrict__ out) {
  size_t idx = (size_t)blockIdx.x * 256 + threadIdx.x;
  if (idx >= (size_t)EE * 160) return;
  int e = (int)(idx / 160);
  int c = (int)(idx - (size_t)e * 160);
  float v = 0.0f;
  if (c < 128) {
    v = (float)h1[(size_t)src[e] * 128 + c];
  } else if (c < 131) {
    int d = c - 128;
    v = posn[(size_t)src[e] * 3 + d] - posn[(size_t)dst[e] * 3 + d];
  }
  out[idx] = (bf16)v;
}

__global__ void __launch_bounds__(256) k_gather3(const bf16* __restrict__ h2,
                                                 const float* __restrict__ posn,
                                                 bf16* __restrict__ out) {
  size_t idx = (size_t)blockIdx.x * 256 + threadIdx.x;
  if (idx >= (size_t)NN * 288) return;
  int n = (int)(idx / 288);
  int c = (int)(idx - (size_t)n * 288);
  float v = 0.0f;
  if (c < 256)      v = (float)h2[(size_t)n * 256 + c];
  else if (c < 259) v = posn[(size_t)n * 3 + (c - 256)];
  out[idx] = (bf16)v;
}

// ---------------------------------------------------------------------------
// Segment max over DEG=16 consecutive edge rows -> node features.
// ---------------------------------------------------------------------------
__global__ void __launch_bounds__(256) k_segmax_edges(const bf16* __restrict__ in,
                                                      bf16* __restrict__ out, int F) {
  size_t idx = (size_t)blockIdx.x * 256 + threadIdx.x;
  size_t total = (size_t)NN * F;
  if (idx >= total) return;
  int f = (int)(idx % F);
  size_t node = idx / F;
  const bf16* base = in + node * (size_t)DEG * F + f;
  float m = -3.4e38f;
#pragma unroll 4
  for (int j = 0; j < DEG; ++j) m = fmaxf(m, (float)base[(size_t)j * F]);
  out[idx] = (bf16)m;
}

// Global max pool: [NN,1024] -> [BG,1024]
__global__ void __launch_bounds__(256) k_pool(const bf16* __restrict__ in,
                                              bf16* __restrict__ out) {
  int idx = blockIdx.x * 256 + threadIdx.x;
  if (idx >= BG * 1024) return;
  int f = idx & 1023, g = idx >> 10;
  const bf16* base = in + (size_t)g * PTS * 1024 + f;
  float m = -3.4e38f;
  for (int p = 0; p < PTS; ++p) m = fmaxf(m, (float)base[(size_t)p * 1024]);
  out[idx] = (bf16)m;
}

// Final regression: [64,512] @ [512,1] + b -> d_out f32 [64]
__global__ void __launch_bounds__(64) k_reg(const bf16* __restrict__ o2,
                                            const float* __restrict__ W,
                                            const float* __restrict__ b,
                                            float* __restrict__ out) {
  int r = threadIdx.x;
  float acc = b[0];
  for (int k = 0; k < 512; ++k) acc += (float)o2[(size_t)r * 512 + k] * W[k];
  out[r] = acc;
}

// ---------------------------------------------------------------------------
// Host launcher
// ---------------------------------------------------------------------------
extern "C" void kernel_launch(void* const* d_in, const int* in_sizes, int n_in,
                              void* d_out, int out_size, void* d_ws, size_t ws_size,
                              hipStream_t stream) {
  (void)in_sizes; (void)n_in; (void)out_size; (void)ws_size;

  // d_in flat order: x, pos, batch, e1_src, e1_dst, e2_src, e2_dst,
  //   mlp1 (3x {W,b,g,be}), mlp2 (3x), mlp3 (3x),
  //   lin1 {W,b}, bn1 {g,be}, lin2 {W,b}, bn2 {g,be}, reg {W,b}
  const float* x    = (const float*)d_in[0];
  const float* pos  = (const float*)d_in[1];
  const int* e1_src = (const int*)d_in[3];
  const int* e1_dst = (const int*)d_in[4];
  const int* e2_src = (const int*)d_in[5];
  const int* e2_dst = (const int*)d_in[6];
  const float* regW = (const float*)d_in[51];
  const float* regb = (const float*)d_in[52];

  struct LD { int w, b, g, be, K, Kp, N; };
  const LD L[11] = {
    { 7,  8,  9, 10,  101,  128,   64},   // mlp1.0
    {11, 12, 13, 14,   64,   64,   64},   // mlp1.1
    {15, 16, 17, 18,   64,   64,  128},   // mlp1.2
    {19, 20, 21, 22,  131,  160,  128},   // mlp2.0
    {23, 24, 25, 26,  128,  128,  128},   // mlp2.1
    {27, 28, 29, 30,  128,  128,  256},   // mlp2.2
    {31, 32, 33, 34,  259,  288,  256},   // mlp3.0
    {35, 36, 37, 38,  256,  256,  512},   // mlp3.1
    {39, 40, 41, 42,  512,  512, 1024},   // mlp3.2
    {43, 44, 45, 46, 1024, 1024,  512},   // lin1 + bn1
    {47, 48, 49, 50,  512,  512,  512},   // lin2 + bn2
  };

  // Workspace carve
  char* ws = (char*)d_ws;
  size_t off = 0;
  auto carve = [&](size_t bytes) -> char* {
    char* p = ws + off;
    off += (bytes + 255) & ~(size_t)255;
    return p;
  };
  float* posn  = (float*)carve((size_t)NN * 3 * 4);
  float* ssum  = (float*)carve(1024 * 4);
  float* ssq   = (float*)carve(1024 * 4);
  bf16*  h1    = (bf16*)carve((size_t)NN * 128 * 2);
  bf16*  h2    = (bf16*)carve((size_t)NN * 256 * 2);
  bf16*  xg    = (bf16*)carve((size_t)BG * 1024 * 2);
  bf16*  o1    = (bf16*)carve((size_t)BG * 512 * 2);
  bf16*  o2    = (bf16*)carve((size_t)BG * 512 * 2);
  size_t packed_elems = 0;
  for (int i = 0; i < 11; ++i) packed_elems += (size_t)L[i].Kp * L[i].N;
  bf16* packed = (bf16*)carve(packed_elems * 2);
  bf16* bufA   = (bf16*)carve((size_t)EE * 160 * 2);  // max width hitting bufA
  bf16* bufB   = (bf16*)carve((size_t)EE * 256 * 2);  // max width hitting bufB

  // Pack all weights (f32 [K,N] -> WMMA fragment order bf16)
  bf16* pw[11];
  { size_t a = 0;
    for (int i = 0; i < 11; ++i) { pw[i] = packed + a; a += (size_t)L[i].Kp * L[i].N; } }
  for (int i = 0; i < 11; ++i) {
    size_t tot = (size_t)L[i].Kp * L[i].N;
    k_pack_w<<<dim3((unsigned)((tot + 255) / 256)), dim3(256), 0, stream>>>(
        (const float*)d_in[L[i].w], pw[i], L[i].K, L[i].Kp, L[i].N);
  }

  auto layer = [&](const bf16* Ain, bf16* Yout, int li, int M,
                   int relu_gemm, int relu_bn) {
    const LD& d = L[li];
    int waves  = (M >> 5) * (d.N >> 6);   // one wave per 32x64 tile
    int blocks = (waves * 32 + 255) / 256;
    k_wmma_gemm<<<dim3(blocks), dim3(256), 0, stream>>>(
        Ain, pw[li], (const float*)d_in[d.b], Yout, M, d.Kp, d.N, relu_gemm);
    k_col_stats<<<dim3(d.N), dim3(256), 0, stream>>>(Yout, ssum, ssq, M, d.N);
    size_t tot = (size_t)M * d.N;
    k_bn_apply<<<dim3((unsigned)((tot + 255) / 256)), dim3(256), 0, stream>>>(
        Yout, ssum, ssq, (const float*)d_in[d.g], (const float*)d_in[d.be],
        M, d.N, relu_bn);
  };

  // NormalizeScale
  k_normalize<<<dim3(BG), dim3(PTS), 0, stream>>>(pos, posn);

  // Stage 1: PointNetConv #1 (edge MLP [101->64->64->128], max-agg)
  { size_t t = (size_t)EE * 128;
    k_gather1<<<dim3((unsigned)((t + 255) / 256)), dim3(256), 0, stream>>>(
        x, posn, e1_src, e1_dst, bufA); }
  layer(bufA, bufB, 0, EE, 1, 0);
  layer(bufB, bufA, 1, EE, 1, 0);
  layer(bufA, bufB, 2, EE, 1, 0);
  { size_t t = (size_t)NN * 128;
    k_segmax_edges<<<dim3((unsigned)((t + 255) / 256)), dim3(256), 0, stream>>>(
        bufB, h1, 128); }

  // Stage 2: PointNetConv #2 (edge MLP [131->128->128->256], max-agg)
  { size_t t = (size_t)EE * 160;
    k_gather2<<<dim3((unsigned)((t + 255) / 256)), dim3(256), 0, stream>>>(
        h1, posn, e2_src, e2_dst, bufA); }
  layer(bufA, bufB, 3, EE, 1, 0);
  layer(bufB, bufA, 4, EE, 1, 0);
  layer(bufA, bufB, 5, EE, 1, 0);
  { size_t t = (size_t)NN * 256;
    k_segmax_edges<<<dim3((unsigned)((t + 255) / 256)), dim3(256), 0, stream>>>(
        bufB, h2, 256); }

  // Stage 3: node MLP [259->256->512->1024]
  { size_t t = (size_t)NN * 288;
    k_gather3<<<dim3((unsigned)((t + 255) / 256)), dim3(256), 0, stream>>>(
        h2, posn, bufA); }
  layer(bufA, bufB, 6, NN, 1, 0);
  layer(bufB, bufA, 7, NN, 1, 0);
  layer(bufA, bufB, 8, NN, 1, 0);

  // Global max pool -> [64,1024]
  { size_t t = (size_t)BG * 1024;
    k_pool<<<dim3((unsigned)((t + 255) / 256)), dim3(256), 0, stream>>>(bufB, xg); }

  // Head: relu(BN(x@W+b)) twice, then regression
  layer(xg, o1,  9, BG, 0, 1);
  layer(o1, o2, 10, BG, 0, 1);
  k_reg<<<dim3(1), dim3(64), 0, stream>>>(o2, regW, regb, (float*)d_out);
}
